// CachingMultiHeadAttention_79242146611282
// MI455X (gfx1250) — compile-verified, matrix operand order checked by reading, not run on hardware
//
#include <hip/hip_runtime.h>
#include <hip/hip_bf16.h>
#include <math.h>

// ---------------- problem constants ----------------
#define B_    2
#define S_    2048
#define DM_   4096
#define NQ_   32
#define NKV_  8
#define HD_   128
#define GROUP_ (NQ_ / NKV_)
#define MROWS_ (B_ * S_)                 // 4096 rows for all GEMMs
#define ATTN_MULT_ 0.08838834764831845f  // 1/sqrt(128)
#define MAX_ATTN_  30.0f

typedef __attribute__((ext_vector_type(16))) __bf16 bf16x16;
typedef __attribute__((ext_vector_type(8)))  __bf16 bf16x8;
typedef __attribute__((ext_vector_type(8)))  float  f32x8;
typedef __attribute__((ext_vector_type(4)))  float  f32x4;

// Load one 16x32 bf16 fragment (A or Bt layout):
// lane<16 takes K {0..7,16..23}, lane>=16 takes K {8..15,24..31};
// caller passes p already offset by row and by kSel (0 or 8).
__device__ __forceinline__ bf16x16 make_frag(const __bf16* p) {
  bf16x8 lo = *(const bf16x8*)(p);
  bf16x8 hi = *(const bf16x8*)(p + 16);
  return __builtin_shufflevector(lo, hi, 0,1,2,3,4,5,6,7,8,9,10,11,12,13,14,15);
}

__device__ __forceinline__ f32x8 wmma_bf16(bf16x16 a, bf16x16 b, f32x8 c) {
  return __builtin_amdgcn_wmma_f32_16x16x32_bf16(
      /*neg_a=*/false, a, /*neg_b=*/false, b,
      /*c_mod=*/(short)0, c, /*reuse_a=*/false, /*reuse_b=*/false);
}

// ---------------- weight transpose + fp32 -> bf16 ----------------
// W (K x N, f32 row-major) -> Wt (N x K, bf16 row-major), 32x32 LDS tiles.
__global__ __launch_bounds__(256)
void transpose_f32_to_bf16(const float* __restrict__ W, __bf16* __restrict__ Wt,
                           int K, int N) {
  __shared__ __bf16 tile[32][33];
  const int k0 = blockIdx.y * 32, n0 = blockIdx.x * 32;
  const int tx = threadIdx.x & 31, ty = threadIdx.x >> 5;   // 32 x 8
#pragma unroll
  for (int r = ty; r < 32; r += 8)
    tile[r][tx] = (__bf16)W[(size_t)(k0 + r) * N + n0 + tx];
  __syncthreads();
#pragma unroll
  for (int r = ty; r < 32; r += 8)
    Wt[(size_t)(n0 + r) * K + k0 + tx] = tile[tx][r];
}

// ---------------- GEMM: C(f32, MxN) = A(f32, MxK) * Bt(bf16, NxK) -----------
// 128x128 block tile, 8 waves, wave tile 32x64, K-step 32, bf16 WMMA compute.
// A: fp32 -> bf16 on the LDS-store path (needs VALU convert anyway).
// B: pre-transposed weights, staged via async global->LDS (ASYNCcnt path).
#define BLK  128
#define BK   32
#define LDSS 40   // 80-byte LDS row stride for sA: 16B aligned

template <int N, int K>
__global__ __launch_bounds__(256)
void gemm_f32a_bf16bt(const float* __restrict__ A, const __bf16* __restrict__ Bt,
                      float* __restrict__ C) {
  __shared__ __bf16 sA[BLK][LDSS];   // sA[m][k]
  __shared__ __bf16 sB[BLK][BK];     // sB[n][k], 64B rows (async copy target)
  const int tid  = threadIdx.x;
  const int lane = tid & 31;
  const int wave = tid >> 5;
  const int blockM = blockIdx.y * BLK;
  const int blockN = blockIdx.x * BLK;
  const int wm = (wave >> 1) * 32;     // wave row offset
  const int wn = (wave & 1) * 64;      // wave col offset
  const int kSel = (lane & 16) ? 8 : 0;
  const int lr   = lane & 15;

  f32x8 acc[2][4];
#pragma unroll
  for (int i = 0; i < 2; i++)
#pragma unroll
    for (int j = 0; j < 4; j++)
#pragma unroll
      for (int r = 0; r < 8; r++) acc[i][j][r] = 0.f;

  const int aRow = tid >> 1, aCol = (tid & 1) * 16;  // A tile: 128 x 32
  const int bRow = tid >> 1, bCol = (tid & 1) * 16;  // B tile: 128(n) x 32(k)
  const unsigned ldsB = (unsigned)(uintptr_t)(&sB[bRow][bCol]);
  const float*  aPtr = A  + (size_t)(blockM + aRow) * K + aCol;
  const __bf16* bPtr = Bt + (size_t)(blockN + bRow) * K + bCol;

  for (int k0 = 0; k0 < K; k0 += BK) {
    const f32x4* ap = (const f32x4*)(aPtr + k0);
    f32x4 a0 = ap[0], a1 = ap[1], a2 = ap[2], a3 = ap[3];
    if (k0 + BK < K)  // prefetch next A tile chunk (global_prefetch_b8)
      __builtin_prefetch(aPtr + k0 + BK, 0, 1);

    __syncthreads();   // previous iteration's fragment reads complete

    // B tile: async global -> LDS copy, 2 x 16B per thread (ASYNCcnt)
    unsigned long long gB = (unsigned long long)(uintptr_t)(bPtr + k0);
    asm volatile("global_load_async_to_lds_b128 %0, %1, off"
                 :: "v"(ldsB), "v"(gB) : "memory");
    asm volatile("global_load_async_to_lds_b128 %0, %1, off offset:16"
                 :: "v"(ldsB), "v"(gB) : "memory");

    // A tile: convert f32 -> bf16, store to LDS
    bf16x8 p0, p1;
#pragma unroll
    for (int e = 0; e < 4; e++) {
      p0[e]   = (__bf16)a0[e];  p0[4+e] = (__bf16)a1[e];
      p1[e]   = (__bf16)a2[e];  p1[4+e] = (__bf16)a3[e];
    }
    *(bf16x8*)(&sA[aRow][aCol])     = p0;
    *(bf16x8*)(&sA[aRow][aCol + 8]) = p1;

    asm volatile("s_wait_asynccnt 0" ::: "memory");  // own async copies done
    __syncthreads();                                 // all waves' tiles visible

    bf16x16 afr[2], bfr[4];
#pragma unroll
    for (int i = 0; i < 2; i++) afr[i] = make_frag(&sA[wm + i*16 + lr][kSel]);
#pragma unroll
    for (int j = 0; j < 4; j++) bfr[j] = make_frag(&sB[wn + j*16 + lr][kSel]);
#pragma unroll
    for (int i = 0; i < 2; i++)
#pragma unroll
      for (int j = 0; j < 4; j++)
        acc[i][j] = wmma_bf16(afr[i], bfr[j], acc[i][j]);
  }

  const int rSel = (lane & 16) ? 8 : 0;   // C layout: VGPR r -> row r (+8 upper half)
#pragma unroll
  for (int i = 0; i < 2; i++)
#pragma unroll
    for (int j = 0; j < 4; j++) {
      float* cp = C + (size_t)(blockM + wm + i*16 + rSel) * N
                    + (blockN + wn + j*16 + lr);
#pragma unroll
      for (int r = 0; r < 8; r++)            // constexpr N -> immediate offsets
        cp[(size_t)r * N] = acc[i][j][r];
    }
}

// ---------------- RoPE + layout kernels ----------------
// q: (B,S,NQ*HD) f32 -> head-major (B,NQ,S,HD) bf16 with RoPE applied
__global__ void rope_q_kernel(const float* __restrict__ qh, __bf16* __restrict__ qbf) {
  int i = blockIdx.x * blockDim.x + threadIdx.x;   // B*S*NQ*64 = 2^23 threads
  int d2 = i & 63;
  int h  = (i >> 6)  & (NQ_ - 1);
  int s  = (i >> 11) & (S_ - 1);
  int b  = i >> 22;
  float inv_freq = __expf(-(float)d2 * (9.210340371976184f / 64.0f)); // 10000^(-2j/128)
  float ph = (float)s * inv_freq;
  float cs = cosf(ph), sn = sinf(ph);
  const float* src = qh + (size_t)(b * S_ + s) * (NQ_ * HD_) + h * HD_;
  float x1 = src[d2], x2 = src[d2 + 64];
  __bf16* dst = qbf + ((size_t)(b * NQ_ + h) * S_ + s) * HD_;
  dst[d2]      = (__bf16)(x1 * cs - x2 * sn);
  dst[d2 + 64] = (__bf16)(x2 * cs + x1 * sn);
}

// k: RoPE, emit cache_k (f32 head-major, part of d_out) + bf16 head-major copy
__global__ void rope_k_kernel(const float* __restrict__ kh, __bf16* __restrict__ kbf,
                              float* __restrict__ cache_k) {
  int i = blockIdx.x * blockDim.x + threadIdx.x;   // B*S*NKV*64 = 2^21 threads
  int d2 = i & 63;
  int h  = (i >> 6) & (NKV_ - 1);
  int s  = (i >> 9) & (S_ - 1);
  int b  = i >> 20;
  float inv_freq = __expf(-(float)d2 * (9.210340371976184f / 64.0f));
  float ph = (float)s * inv_freq;
  float cs = cosf(ph), sn = sinf(ph);
  const float* src = kh + (size_t)(b * S_ + s) * (NKV_ * HD_) + h * HD_;
  float x1 = src[d2], x2 = src[d2 + 64];
  float o1 = x1 * cs - x2 * sn;
  float o2 = x2 * cs + x1 * sn;
  size_t base = ((size_t)(b * NKV_ + h) * S_ + s) * HD_;
  cache_k[base + d2]      = o1;
  cache_k[base + d2 + 64] = o2;
  kbf[base + d2]      = (__bf16)o1;
  kbf[base + d2 + 64] = (__bf16)o2;
}

// v: emit cache_v (f32 head-major) + bf16 transposed (B,NKV,HD,S) for PV WMMA
__global__ void conv_v_kernel(const float* __restrict__ vh, __bf16* __restrict__ vT,
                              float* __restrict__ cache_v) {
  int i = blockIdx.x * blockDim.x + threadIdx.x;   // B*S*NKV*HD = 2^22 threads
  int d = i & 127;
  int h = (i >> 7)  & (NKV_ - 1);
  int s = (i >> 10) & (S_ - 1);
  int b = i >> 21;
  float x = vh[(size_t)(b * S_ + s) * (NKV_ * HD_) + h * HD_ + d];
  cache_v[((size_t)(b * NKV_ + h) * S_ + s) * HD_ + d] = x;
  vT[((size_t)(b * NKV_ + h) * HD_ + d) * S_ + s] = (__bf16)x;
}

// ---------------- flash attention (one wave per 16-query tile) --------------
__global__ __launch_bounds__(256)
void attention_kernel(const __bf16* __restrict__ Qh,   // (B,NQ,S,HD)
                      const __bf16* __restrict__ Kh,   // (B,NKV,S,HD)
                      const __bf16* __restrict__ VT,   // (B,NKV,HD,S)
                      float* __restrict__ attnOut) {   // (B,S,NQ*HD)
  __shared__ __bf16 sP[8][16][32];   // per-wave P scratch (C-layout -> A-layout shim)
  const int lane = threadIdx.x & 31;
  const int wave = threadIdx.x >> 5;
  const int idx  = blockIdx.x * 8 + wave;       // (b, qhead, qtile)
  const int QT   = S_ / 16;
  const int qt   = idx % QT;
  const int t2   = idx / QT;
  const int head = t2 % NQ_;
  const int b    = t2 / NQ_;
  const int kvh  = head / GROUP_;
  const int qBase = qt * 16;
  const int lr   = lane & 15;
  const int kSel = (lane & 16) ? 8 : 0;
  const int rSel = (lane & 16) ? 8 : 0;

  const __bf16* Qb = Qh + ((size_t)(b * NQ_  + head) * S_) * HD_;
  const __bf16* Kb = Kh + ((size_t)(b * NKV_ + kvh ) * S_) * HD_;
  const __bf16* Vb = VT + ((size_t)(b * NKV_ + kvh ) * HD_) * S_;

  bf16x16 qf[4];   // Q tile 16x128 as 4 A-fragments over HD
#pragma unroll
  for (int c = 0; c < 4; c++)
    qf[c] = make_frag(Qb + (size_t)(qBase + lr) * HD_ + c * 32 + kSel);

  f32x8 o[8];      // O tile 16x128
#pragma unroll
  for (int j = 0; j < 8; j++)
#pragma unroll
    for (int r = 0; r < 8; r++) o[j][r] = 0.f;
  float mrow[8], lsum[8];
#pragma unroll
  for (int r = 0; r < 8; r++) { mrow[r] = -3.0e38f; lsum[r] = 0.f; }

  const int kvEnd = qBase + 16;                  // causal: keys <= qBase+15
  for (int kb = 0; kb < kvEnd; kb += 32) {
    f32x8 s0, s1;                                // logits for keys kb..kb+15, +16..+31
#pragma unroll
    for (int r = 0; r < 8; r++) { s0[r] = 0.f; s1[r] = 0.f; }
#pragma unroll
    for (int c = 0; c < 4; c++) {                // contract over HD (4 x K=32)
      bf16x16 kf0 = make_frag(Kb + (size_t)(kb      + lr) * HD_ + c * 32 + kSel);
      bf16x16 kf1 = make_frag(Kb + (size_t)(kb + 16 + lr) * HD_ + c * 32 + kSel);
      s0 = wmma_bf16(qf[c], kf0, s0);
      s1 = wmma_bf16(qf[c], kf1, s1);
    }
    const bool needMask = (kb + 31 > qBase);

#pragma unroll
    for (int r = 0; r < 8; r++) {                // online softmax, per row
      int rowG = qBase + r + rSel;
      float x0 = MAX_ATTN_ * tanhf(s0[r] * ATTN_MULT_ * (1.0f / MAX_ATTN_));
      float x1 = MAX_ATTN_ * tanhf(s1[r] * ATTN_MULT_ * (1.0f / MAX_ATTN_));
      if (needMask) {
        if (kb + lr      > rowG) x0 = -1.0e30f;
        if (kb + 16 + lr > rowG) x1 = -1.0e30f;
      }
      float mx = fmaxf(x0, x1);                  // reduce across 16-lane half
#pragma unroll
      for (int d = 1; d < 16; d <<= 1) mx = fmaxf(mx, __shfl_xor(mx, d, 32));
      float mnew  = fmaxf(mrow[r], mx);
      float alpha = __expf(mrow[r] - mnew);
      mrow[r] = mnew;
      float p0 = __expf(x0 - mnew);
      float p1 = __expf(x1 - mnew);
      float ps = p0 + p1;
#pragma unroll
      for (int d = 1; d < 16; d <<= 1) ps += __shfl_xor(ps, d, 32);
      lsum[r] = lsum[r] * alpha + ps;
#pragma unroll
      for (int j = 0; j < 8; j++) o[j][r] *= alpha;
      int rt = r + rSel;                         // row within tile
      sP[wave][rt][lr]      = (__bf16)p0;
      sP[wave][rt][lr + 16] = (__bf16)p1;
    }
    asm volatile("s_wait_dscnt 0" ::: "memory"); // wave-local LDS RAW guard
    bf16x16 pf = make_frag(&sP[wave][lr][kSel]); // P as A-fragment (16x32)
#pragma unroll
    for (int j = 0; j < 8; j++) {                // O += P @ V  (contract 32 keys)
      bf16x16 vf = make_frag(Vb + (size_t)(j * 16 + lr) * S_ + kb + kSel);
      o[j] = wmma_bf16(pf, vf, o[j]);
    }
  }

  float inv[8];
#pragma unroll
  for (int r = 0; r < 8; r++) inv[r] = 1.0f / lsum[r];
#pragma unroll
  for (int j = 0; j < 8; j++)
#pragma unroll
    for (int r = 0; r < 8; r++) {
      int s  = qBase + r + rSel;
      int hd = j * 16 + lr;
      attnOut[((size_t)(b * S_ + s) * NQ_ + head) * HD_ + hd] = o[j][r] * inv[r];
    }
}

// ---------------- host orchestration ----------------
extern "C" void kernel_launch(void* const* d_in, const int* in_sizes, int n_in,
                              void* d_out, int out_size, void* d_ws, size_t ws_size,
                              hipStream_t stream) {
  const float* query = (const float*)d_in[0];
  const float* key   = (const float*)d_in[1];
  const float* value = (const float*)d_in[2];
  // d_in[3] = mask: causal-tril, applied analytically in the attention kernel
  const float* Wq = (const float*)d_in[4];
  const float* Wk = (const float*)d_in[5];
  const float* Wv = (const float*)d_in[6];
  const float* Wo = (const float*)d_in[7];

  float* out     = (float*)d_out;                       // (B,S,DM)
  float* cache_k = out + (size_t)B_ * S_ * DM_;         // (B,NKV,S,HD)
  float* cache_v = cache_k + (size_t)B_ * NKV_ * S_ * HD_;

  char* ws = (char*)d_ws;
  size_t off = 0;
  auto bump = [&](size_t bytes) -> void* {
    void* p = ws + off;
    off += (bytes + 255) & ~(size_t)255;
    return p;
  };
  __bf16* Wqt = (__bf16*)bump((size_t)DM_ * DM_ * 2);          // 32 MB (N x K)
  __bf16* Wkt = (__bf16*)bump((size_t)DM_ * NKV_ * HD_ * 2);   //  8 MB
  __bf16* Wvt = (__bf16*)bump((size_t)DM_ * NKV_ * HD_ * 2);   //  8 MB
  __bf16* Wot = (__bf16*)bump((size_t)DM_ * DM_ * 2);          // 32 MB
  float*  qh  = (float*)bump((size_t)MROWS_ * DM_ * 4);        // 64 MB (reused as attn)
  float*  kh  = (float*)bump((size_t)MROWS_ * NKV_ * HD_ * 4); // 16 MB
  float*  vh  = (float*)bump((size_t)MROWS_ * NKV_ * HD_ * 4); // 16 MB
  __bf16* qbf = (__bf16*)bump((size_t)B_ * NQ_ * S_ * HD_ * 2);  // 32 MB
  __bf16* kbf = (__bf16*)bump((size_t)B_ * NKV_ * S_ * HD_ * 2); //  8 MB
  __bf16* vT  = (__bf16*)bump((size_t)B_ * NKV_ * HD_ * S_ * 2); //  8 MB
  float* attnf = qh;  // qh is dead after rope_q: alias attention output onto it

  // 1) weights -> bf16, transposed to (N, K) so GEMM B staging is linear
  dim3 tb(256);
  transpose_f32_to_bf16<<<dim3(DM_/32, DM_/32), tb, 0, stream>>>(Wq, Wqt, DM_, DM_);
  transpose_f32_to_bf16<<<dim3((NKV_*HD_)/32, DM_/32), tb, 0, stream>>>(Wk, Wkt, DM_, NKV_*HD_);
  transpose_f32_to_bf16<<<dim3((NKV_*HD_)/32, DM_/32), tb, 0, stream>>>(Wv, Wvt, DM_, NKV_*HD_);
  transpose_f32_to_bf16<<<dim3(DM_/32, DM_/32), tb, 0, stream>>>(Wo, Wot, DM_, DM_);

  // 2) QKV projections (WMMA, async-LDS B staging)
  dim3 blk(256);
  dim3 gq(DM_ / BLK, MROWS_ / BLK);               // 32 x 32
  dim3 gkv((NKV_ * HD_) / BLK, MROWS_ / BLK);     //  8 x 32
  gemm_f32a_bf16bt<DM_,        DM_><<<gq,  blk, 0, stream>>>(query, Wqt, qh);
  gemm_f32a_bf16bt<NKV_ * HD_, DM_><<<gkv, blk, 0, stream>>>(key,   Wkt, kh);
  gemm_f32a_bf16bt<NKV_ * HD_, DM_><<<gkv, blk, 0, stream>>>(value, Wvt, vh);

  // 3) RoPE + layout shuffles + cache outputs
  rope_q_kernel<<<(B_ * S_ * NQ_  * 64)  / 256, 256, 0, stream>>>(qh, qbf);
  rope_k_kernel<<<(B_ * S_ * NKV_ * 64)  / 256, 256, 0, stream>>>(kh, kbf, cache_k);
  conv_v_kernel<<<(B_ * S_ * NKV_ * HD_) / 256, 256, 0, stream>>>(vh, vT, cache_v);

  // 4) causal GQA flash attention (WMMA), one wave per 16-row query tile
  attention_kernel<<<(B_ * NQ_ * (S_ / 16)) / 8, 256, 0, stream>>>(qbf, kbf, vT, attnf);

  // 5) output projection (WMMA) straight into d_out
  gemm_f32a_bf16bt<DM_, DM_><<<gq, blk, 0, stream>>>(attnf, Wot, out);

  (void)in_sizes; (void)n_in; (void)out_size; (void)ws_size;
}
